// Spatial_Attention_34522947125744
// MI455X (gfx1250) — matrix operand order repaired and checked by reading, not compiled, not run on hardware
//
#include <hip/hip_runtime.h>

typedef __attribute__((ext_vector_type(16))) _Float16 v16h;
typedef __attribute__((ext_vector_type(8)))  _Float16 v8h;
typedef __attribute__((ext_vector_type(8)))  float    v8f;
typedef __attribute__((ext_vector_type(4)))  float    v4f;

static __device__ __forceinline__ v16h cat8(v8h lo, v8h hi) {
  return __builtin_shufflevector(lo, hi, 0, 1, 2, 3, 4, 5, 6, 7,
                                 8, 9, 10, 11, 12, 13, 14, 15);
}

// gfx1250 async global->LDS copy, 16B per lane. Tracked by ASYNCcnt.
// LDS offset = low 32 bits of the flat shared-aperture address (ISA 10.2).
static __device__ __forceinline__ void async_ld_b128(void* lds_ptr, const void* gptr) {
  unsigned lds = (unsigned)(size_t)lds_ptr;
  asm volatile("global_load_async_to_lds_b128 %0, %1, off"
               :
               : "v"(lds), "v"(gptr)
               : "memory");
}
static __device__ __forceinline__ void wait_async0() {
  asm volatile("s_wait_asynccnt 0x0" ::: "memory");
}

// ---------------- weight packing (transposed: Wt[n][k]) ----------------

__global__ __launch_bounds__(256) void f32_to_f16_kernel(const float* __restrict__ in,
                                                         _Float16* __restrict__ out, int n) {
  int i = blockIdx.x * blockDim.x + threadIdx.x;
  if (i < n) out[i] = (_Float16)in[i];
}

// WallT[n][k] (576 x 192 f16): n<192 -> Wq[k][n], else Wkv[k][n-192]
__global__ __launch_bounds__(256) void pack_wallT_kernel(const float* __restrict__ Wq,
                                                         const float* __restrict__ Wkv,
                                                         _Float16* __restrict__ WallT) {
  int i = blockIdx.x * blockDim.x + threadIdx.x;
  const int total = 576 * 192;
  if (i >= total) return;
  int n = i / 192;
  int k = i % 192;
  float v = (n < 192) ? Wq[k * 192 + n] : Wkv[k * 384 + (n - 192)];
  WallT[i] = (_Float16)v;
}

// WprojT[n][k] (192 x 192 f16) = Wproj[k][n]
__global__ __launch_bounds__(256) void pack_wprojT_kernel(const float* __restrict__ Wp,
                                                          _Float16* __restrict__ WpT) {
  int i = blockIdx.x * blockDim.x + threadIdx.x;
  const int total = 192 * 192;
  if (i >= total) return;
  int n = i / 192;
  int k = i % 192;
  WpT[i] = (_Float16)Wp[k * 192 + n];
}

// ---------------- f16 WMMA GEMM ----------------
// Block tile 128x64, 8 waves, K-step 32, double-buffered async A staging.
// A fragments: 2x ds_load_b128 from LDS (row stride 40 halves = 80B).
// B fragments: 2x global_load_b128 from pre-transposed Wt[n][k] (L2-resident).

#define TM 128
#define TN 64
#define TK 32
#define LDS_STRIDE 40   // halves; 80 bytes, multiple of 16

template <bool OUTF32>
__device__ __forceinline__ void gemm_body(const _Float16* __restrict__ A,
                                          const _Float16* __restrict__ Wt,  // [N][K]
                                          const float* __restrict__ bias,
                                          void* __restrict__ Cout,
                                          int M, int N, int K) {
  __shared__ __align__(16) _Float16 As[2][TM][LDS_STRIDE];

  int tid  = threadIdx.x;
  int lane = tid & 31;
  int wave = tid >> 5;
  int l15  = lane & 15;
  int m0 = blockIdx.x * TM;
  int n0 = blockIdx.y * TN;
  int wrow = wave * 16;

  auto stageA = [&](int buf, int kk) {
#pragma unroll
    for (int ch = tid; ch < TM * 4; ch += 256) {
      int r = ch >> 2, p = ch & 3;
      async_ld_b128(&As[buf][r][p * 8], &A[(size_t)(m0 + r) * K + kk + p * 8]);
    }
  };

  v8f acc[4] = {};

  stageA(0, 0);
  wait_async0();
  __syncthreads();

  int buf = 0;
  for (int k0 = 0; k0 < K; k0 += TK, buf ^= 1) {
    if (k0 + TK < K) stageA(buf ^ 1, k0 + TK);   // overlap copy of next tile

    // A fragment: lanes 0-15 -> K 0..7 & 16..23 ; lanes 16-31 -> K 8..15 & 24..31
    int arow = wrow + l15;
    int kb  = (lane < 16) ? 0 : 8;
    v16h af = cat8(*(const v8h*)&As[buf][arow][kb],
                   *(const v8h*)&As[buf][arow][kb + 16]);

    int kb2 = (lane < 16) ? 0 : 8;               // B: lanes 0-15 K 0..7 & 16..23; 16-31 K 8..15 & 24..31
#pragma unroll
    for (int nt = 0; nt < 4; ++nt) {
      int bcol = n0 + nt * 16 + l15;
      const _Float16* wr = &Wt[(size_t)bcol * K + k0];
      v16h bf = cat8(*(const v8h*)&wr[kb2], *(const v8h*)&wr[kb2 + 16]);
      acc[nt] = __builtin_amdgcn_wmma_f32_16x16x32_f16(false, af, false, bf,
                                                       (short)0, acc[nt], false, false);
    }

    wait_async0();
    __syncthreads();
  }

#pragma unroll
  for (int nt = 0; nt < 4; ++nt) {
    int col = n0 + nt * 16 + l15;
    float b = OUTF32 ? bias[col] : 0.f;
#pragma unroll
    for (int i = 0; i < 8; ++i) {
      int row = m0 + wrow + ((lane < 16) ? i : 8 + i);
      if (OUTF32)
        ((float*)Cout)[(size_t)row * N + col] = acc[nt][i] + b;
      else
        ((_Float16*)Cout)[(size_t)row * N + col] = (_Float16)acc[nt][i];
    }
  }
}

__global__ __launch_bounds__(256) void gemm_f16f16(const _Float16* __restrict__ A,
                                                   const _Float16* __restrict__ Wt,
                                                   _Float16* __restrict__ C,
                                                   int M, int N, int K) {
  gemm_body<false>(A, Wt, nullptr, C, M, N, K);
}

__global__ __launch_bounds__(256) void gemm_f16f32_bias(const _Float16* __restrict__ A,
                                                        const _Float16* __restrict__ Wt,
                                                        const float* __restrict__ bias,
                                                        float* __restrict__ C,
                                                        int M, int N, int K) {
  gemm_body<true>(A, Wt, bias, C, M, N, K);
}

// ---------------- windowed attention ----------------
// One block (128 threads = 4 waves) per (batch, branch, window); one wave per head.
// N = 32 tokens, head dim 24 (zero-padded to 32 for the WMMA K dimension).

#define SLDF 36  // f32 LDS stride for S (144B, multiple of 16)

__global__ __launch_bounds__(128) void win_attn_kernel(const _Float16* __restrict__ qkv,
                                                       _Float16* __restrict__ out,
                                                       int Hdim, int Wdim) {
  int bid = blockIdx.x;
  int b   = bid >> 10;         // 2*512 windows per batch
  int br  = (bid >> 9) & 1;
  int win = bid & 511;

  int wh = (br == 0) ? 4 : 8;
  int ww = (br == 0) ? 8 : 4;
  int rh = (br == 0) ? 2 : 4;  // roll shift (rows)
  int rw = (br == 0) ? 4 : 2;  // roll shift (cols)
  int sh = wh >> 1, sw = ww >> 1;
  int nv = Wdim / ww;
  int wu = win / nv, wv = win % nv;
  int Cb = br * 96;

  __shared__ __align__(16) _Float16 q_s[4][32][LDS_STRIDE];   // Q, later P
  __shared__ __align__(16) _Float16 k_s[4][32][LDS_STRIDE];   // K: [key][chan]
  __shared__ __align__(16) _Float16 vT_s[4][32][LDS_STRIDE];  // V^T: [chan][key]
  __shared__ __align__(16) float    s_s[4][32][SLDF];
  __shared__ int m_g[32];          // global row per window token
  __shared__ int reg_s[32];        // Swin region id per token (rolled coords)

  int tid  = threadIdx.x;
  int lane = tid & 31;
  int hd   = tid >> 5;
  int l15  = lane & 15;

  if (tid < 32) {
    int hh = tid / ww, wi = tid % ww;
    int h_r = wu * wh + hh;
    int w_r = wv * ww + wi;
    int h_o = (h_r + rh) % Hdim;
    int w_o = (w_r + rw) % Wdim;
    m_g[tid] = b * (Hdim * Wdim) + h_o * Wdim + w_o;
    int hreg = (h_r >= Hdim - wh ? 1 : 0) + (h_r >= Hdim - sh ? 1 : 0);
    int wreg = (w_r >= Wdim - ww ? 1 : 0) + (w_r >= Wdim - sw ? 1 : 0);
    reg_s[tid] = hreg * 3 + wreg;
  }
  // Zero the channel pads while indices are being computed.
  {
    v8h z = {};
    for (int i = tid; i < 4 * 32; i += 128) {
      int h2 = i >> 5, n = i & 31;
      *(v8h*)&q_s[h2][n][24] = z;
      *(v8h*)&k_s[h2][n][24] = z;
      if (n >= 24) {                    // K-pad rows of V^T (chan 24..31)
        *(v8h*)&vT_s[h2][n][0]  = z;
        *(v8h*)&vT_s[h2][n][8]  = z;
        *(v8h*)&vT_s[h2][n][16] = z;
        *(v8h*)&vT_s[h2][n][24] = z;
      }
    }
  }
  __syncthreads();

  // Stage Q/K asynchronously (row-major [token][chan]); V synchronously transposed.
  // 4 heads * 32 tokens * 3 chunks(8ch) = 384 chunks; 3 per thread.
  for (int i = tid; i < 4 * 32 * 3; i += 128) {
    int h2 = i / 96;
    int n  = (i / 3) & 31;
    int p  = i % 3;                      // channel group 0..2 (8 each; 24 total)
    size_t base = (size_t)m_g[n] * 576 + (size_t)(Cb + h2 * 24 + p * 8);
    async_ld_b128(&q_s[h2][n][p * 8], &qkv[base]);
    async_ld_b128(&k_s[h2][n][p * 8], &qkv[base + 192]);
    v8h vv = *(const v8h*)&qkv[base + 384];
#pragma unroll
    for (int j = 0; j < 8; ++j) vT_s[h2][p * 8 + j][n] = vv[j];
  }
  wait_async0();
  __syncthreads();

  // ---- S = Q @ K^T  (2x2 WMMA tiles, K=32 padded channel dim) ----
  v8f sacc[2][2] = {};
  {
    int kb  = (lane < 16) ? 0 : 8;
    int kb2 = (lane < 16) ? 0 : 8;
#pragma unroll
    for (int mi = 0; mi < 2; ++mi) {
      int arow = mi * 16 + l15;
      v16h af = cat8(*(const v8h*)&q_s[hd][arow][kb],
                     *(const v8h*)&q_s[hd][arow][kb + 16]);
#pragma unroll
      for (int ni = 0; ni < 2; ++ni) {
        int bn = ni * 16 + l15;          // key index (column of S)
        v16h bf = cat8(*(const v8h*)&k_s[hd][bn][kb2],
                       *(const v8h*)&k_s[hd][bn][kb2 + 16]);
        sacc[mi][ni] = __builtin_amdgcn_wmma_f32_16x16x32_f16(false, af, false, bf,
                                                              (short)0, sacc[mi][ni],
                                                              false, false);
      }
    }
  }
  // spill S to LDS
#pragma unroll
  for (int mi = 0; mi < 2; ++mi)
#pragma unroll
    for (int ni = 0; ni < 2; ++ni)
#pragma unroll
      for (int i = 0; i < 8; ++i) {
        int row = mi * 16 + ((lane < 16) ? i : 8 + i);
        int col = ni * 16 + l15;
        s_s[hd][row][col] = sacc[mi][ni][i];
      }
  __syncthreads();

  // ---- masked softmax, one thread per (head,row); P -> q_s ----
  {
    int h2 = tid >> 5, row = tid & 31;
    const float scale = 0.2041241452319315f;   // 24^-0.5
    int ri = reg_s[row];
    float vals[32];
    float mx = -1e30f;
#pragma unroll
    for (int g = 0; g < 8; ++g) {
      v4f sv = *(const v4f*)&s_s[h2][row][g * 4];
#pragma unroll
      for (int j = 0; j < 4; ++j) {
        int c = g * 4 + j;
        float m = (reg_s[c] != ri) ? -100.0f : 0.0f;
        float s = sv[j] * scale + m;
        vals[c] = s;
        mx = fmaxf(mx, s);
      }
    }
    float sum = 0.f;
#pragma unroll
    for (int j = 0; j < 32; ++j) {
      float e = __expf(vals[j] - mx);
      vals[j] = e;
      sum += e;
    }
    float inv = 1.0f / sum;
#pragma unroll
    for (int g = 0; g < 4; ++g) {
      v8h pv;
#pragma unroll
      for (int j = 0; j < 8; ++j) pv[j] = (_Float16)(vals[g * 8 + j] * inv);
      *(v8h*)&q_s[h2][row][g * 8] = pv;
    }
  }
  __syncthreads();

  // ---- O = P @ V  (K = 32 keys; B = vT[chan][key] contiguous per lane) ----
  v8f oacc[2][2] = {};
  {
    int kb  = (lane < 16) ? 0 : 8;
    int kb2 = (lane < 16) ? 0 : 8;
#pragma unroll
    for (int mi = 0; mi < 2; ++mi) {
      int arow = mi * 16 + l15;
      v16h af = cat8(*(const v8h*)&q_s[hd][arow][kb],
                     *(const v8h*)&q_s[hd][arow][kb + 16]);
#pragma unroll
      for (int ni = 0; ni < 2; ++ni) {
        int bn = ni * 16 + l15;          // output channel
        v16h bf = cat8(*(const v8h*)&vT_s[hd][bn][kb2],
                       *(const v8h*)&vT_s[hd][bn][kb2 + 16]);
        oacc[mi][ni] = __builtin_amdgcn_wmma_f32_16x16x32_f16(false, af, false, bf,
                                                              (short)0, oacc[mi][ni],
                                                              false, false);
      }
    }
  }
  // scatter O back to (b, hw, channel) at original coordinates
#pragma unroll
  for (int mi = 0; mi < 2; ++mi)
#pragma unroll
    for (int ni = 0; ni < 2; ++ni)
#pragma unroll
      for (int i = 0; i < 8; ++i) {
        int n  = mi * 16 + ((lane < 16) ? i : 8 + i);
        int cc = ni * 16 + l15;
        if (cc < 24)
          out[(size_t)m_g[n] * 192 + (Cb + hd * 24 + cc)] = (_Float16)oacc[mi][ni][i];
      }
}

// ---------------- host-side launch ----------------

extern "C" void kernel_launch(void* const* d_in, const int* in_sizes, int n_in,
                              void* d_out, int out_size, void* d_ws, size_t ws_size,
                              hipStream_t stream) {
  (void)in_sizes; (void)n_in; (void)out_size; (void)ws_size;

  const float* x     = (const float*)d_in[0];   // (8, 16384, 192)
  const float* Wq    = (const float*)d_in[1];   // (192, 192)
  const float* Wkv   = (const float*)d_in[2];   // (192, 384)
  const float* Wproj = (const float*)d_in[3];   // (192, 192)
  const float* bproj = (const float*)d_in[4];   // (192,)
  float* outp = (float*)d_out;

  const int B = 8, Hd = 128, Wd = 128, C = 192;
  const int M = B * Hd * Wd;                    // 131072
  const int Nqkv = 576;

  char* ws = (char*)d_ws;
  size_t off = 0;
  _Float16* xh     = (_Float16*)(ws + off); off += (size_t)M * C * 2;
  _Float16* wallT  = (_Float16*)(ws + off); off += (size_t)Nqkv * C * 2;
  _Float16* wprjT  = (_Float16*)(ws + off); off += (size_t)C * C * 2;
  _Float16* qkv    = (_Float16*)(ws + off); off += (size_t)M * Nqkv * 2;
  _Float16* attn   = (_Float16*)(ws + off); off += (size_t)M * C * 2;

  // 1) converts / weight transposes
  {
    int n = M * C;
    f32_to_f16_kernel<<<(n + 255) / 256, 256, 0, stream>>>(x, xh, n);
    int nw = Nqkv * C;
    pack_wallT_kernel<<<(nw + 255) / 256, 256, 0, stream>>>(Wq, Wkv, wallT);
    int np = C * C;
    pack_wprojT_kernel<<<(np + 255) / 256, 256, 0, stream>>>(Wproj, wprjT);
  }

  // 2) QKV projection: (M x 192) @ (192 x 576) -> f16
  {
    dim3 grid(M / TM, Nqkv / TN);
    gemm_f16f16<<<grid, 256, 0, stream>>>(xh, wallT, qkv, M, Nqkv, C);
  }

  // 3) windowed attention: 8 batches x 2 branches x 512 windows
  {
    dim3 grid(B * 2 * 512);
    win_attn_kernel<<<grid, 128, 0, stream>>>(qkv, attn, Hd, Wd);
  }

  // 4) output projection: (M x 192) @ (192 x 192) + bias -> f32
  {
    dim3 grid(M / TM, C / TN);
    gemm_f16f32_bias<<<grid, 256, 0, stream>>>(attn, wprjT, bproj, outp, M, C, C);
  }
}
